// xLSTMMoELayer_56813827391691
// MI455X (gfx1250) — matrix operand order repaired and checked by reading, not compiled, not run on hardware
//
#include <hip/hip_runtime.h>
#include <hip/hip_bf16.h>

// MoE (top-2 of 8 experts), T=8192 tokens, D=1024, F=4096.
// Sparse dispatch + bf16 WMMA (f32 accumulate), fused branch-free gelu,
// h kept in LDS, x tile gathered via the Tensor Data Mover (gather mode).

#define T_TOK   8192
#define D_DIM   1024
#define F_DIM   4096
#define N_EXP   8

typedef __bf16 bf16_t;
typedef __attribute__((ext_vector_type(16))) __bf16 v16bf;
typedef __attribute__((ext_vector_type(8)))  __bf16 v8bf;
typedef __attribute__((ext_vector_type(8)))  float  v8f;
typedef __attribute__((ext_vector_type(4))) unsigned int u32x4;
typedef __attribute__((ext_vector_type(8))) int          i32x8;
typedef __attribute__((ext_vector_type(4))) int          i32x4;

// LDS layout (bytes). xs_f32 aliases the upper half of hs: it is only live
// before hs is first written (GEMM1 epilogue comes after the conversion).
#define HS_OFF      0                   // bf16 [16][4096] = 128 KB
#define XF32_OFF    65536               // f32  [16][1024] =  64 KB (alias)
#define XBF_OFF     131072              // bf16 [16][1024] =  32 KB
#define TK_OFF      163840              // 16 int + 16 float
#define SMEM_BYTES  (163840 + 128)

// ---------------------------------------------------------------- helpers

__device__ __forceinline__ float gelu_tanh(float x) {
    // jax.nn.gelu tanh approximation, branch-free tanh:
    // tanh(y) = sign(y) * (1 - 2e/(1+e)),  e = exp(-2|y|)  (e <= 1, no overflow)
    const float y  = 0.7978845608028654f * (x + 0.044715f * x * x * x);
    const float e  = __expf(-2.0f * fabsf(y));
    const float t  = 1.0f - 2.0f * e * __builtin_amdgcn_rcpf(1.0f + e);
    return 0.5f * x * (1.0f + __builtin_copysignf(t, y));
}

// A-fragment (16x32 bf16, row-major source) from LDS.
// contiguous k order used here (not the ISA table order): lanes 0-15 -> M=lane, K = 0..15;
// lanes 16-31 -> M=lane-16, K = 16..31.
__device__ __forceinline__ v16bf load_a_frag(const bf16_t* __restrict__ base,
                                             int ld, int k0, int lane) {
    const int row  = lane & 15;
    const int half = lane >> 4;
    const bf16_t* p = base + row * ld + k0 + half * 16;
    v8bf lo = *(const v8bf*)(p);
    v8bf hi = *(const v8bf*)(p + 8);
    v16bf a;
#pragma unroll
    for (int i = 0; i < 8; ++i) { a[i] = lo[i]; a[i + 8] = hi[i]; }
    return a;
}

// Two adjacent B-fragments (32x16 each, cols [n0,n0+16) and [n0+16,n0+32))
// from global f32 weights; per weight row the wave touches a full 128B line.
// Lanes 0-15 -> N=lane, K=k0+0..15; lanes 16-31 -> N=lane-16, K=k0+16..31.
__device__ __forceinline__ void load_b_pair_f32(const float* __restrict__ w,
                                                int ld, int k0, int n0, int lane,
                                                v16bf& b0, v16bf& b1) {
    const int col = n0 + (lane & 15);
    const int kb  = k0 + (lane >> 4) * 16;
    const float* p = w + (size_t)kb * ld + col;
#pragma unroll
    for (int i = 0; i < 16; ++i) {
        b0[i] = (bf16_t)p[(size_t)i * ld];
        b1[i] = (bf16_t)p[(size_t)i * ld + 16];
    }
}

// ---------------------------------------------------------------- kernels

__global__ void moe_init_kernel(float* __restrict__ out, size_t n,
                                int* __restrict__ counts) {
    size_t i = (size_t)blockIdx.x * blockDim.x + threadIdx.x;
    size_t step = (size_t)gridDim.x * blockDim.x;
    for (; i < n; i += step) out[i] = 0.0f;
    if (blockIdx.x == 0 && threadIdx.x < N_EXP) counts[threadIdx.x] = 0;
}

// One wave32 per token: logits = x . gate_w, softmax, top-2, append to lists.
__global__ void moe_router_kernel(const float* __restrict__ x,
                                  const float* __restrict__ gw,
                                  int* __restrict__ counts,
                                  int* __restrict__ tokids,
                                  float* __restrict__ tokw) {
    const int wave = threadIdx.x >> 5;
    const int lane = threadIdx.x & 31;
    const int t = blockIdx.x * 8 + wave;
    if (t >= T_TOK) return;

    float acc[N_EXP];
#pragma unroll
    for (int e = 0; e < N_EXP; ++e) acc[e] = 0.0f;

    const float* xr = x + (size_t)t * D_DIM;
    for (int d = lane; d < D_DIM; d += 32) {
        const float xv = xr[d];
        const float* g = gw + (size_t)d * N_EXP;
#pragma unroll
        for (int e = 0; e < N_EXP; ++e) acc[e] += xv * g[e];
    }
#pragma unroll
    for (int e = 0; e < N_EXP; ++e) {
#pragma unroll
        for (int off = 16; off > 0; off >>= 1)
            acc[e] += __shfl_xor(acc[e], off, 32);
    }

    if (lane == 0) {
        float m = acc[0];
#pragma unroll
        for (int e = 1; e < N_EXP; ++e) m = fmaxf(m, acc[e]);
        float p[N_EXP];
#pragma unroll
        for (int e = 0; e < N_EXP; ++e) p[e] = __expf(acc[e] - m);
        int e0 = 0;
#pragma unroll
        for (int e = 1; e < N_EXP; ++e) if (p[e] > p[e0]) e0 = e;
        int e1 = (e0 == 0) ? 1 : 0;
#pragma unroll
        for (int e = 0; e < N_EXP; ++e)
            if (e != e0 && p[e] > p[e1]) e1 = e;
        const float inv = 1.0f / (p[e0] + p[e1]);
        const float w0 = p[e0] * inv, w1 = p[e1] * inv;

        int s0 = atomicAdd(&counts[e0], 1);
        tokids[e0 * T_TOK + s0] = t;
        tokw  [e0 * T_TOK + s0] = w0;
        int s1 = atomicAdd(&counts[e1], 1);
        tokids[e1 * T_TOK + s1] = t;
        tokw  [e1 * T_TOK + s1] = w1;
    }
}

// Fused expert FFN: block = (tile of 16 tokens, expert e), 256 threads = 8 waves.
// x rows gathered into LDS by the TDM (gather mode, 16 row indices), then
// GEMM1 -> +b1 -> gelu -> h (bf16 LDS) -> GEMM2 -> +b2 -> *weight -> atomic add.
__global__ void moe_expert_ffn_kernel(const float* __restrict__ x,
                                      const float* __restrict__ w1,
                                      const float* __restrict__ b1,
                                      const float* __restrict__ w2,
                                      const float* __restrict__ b2,
                                      const int*   __restrict__ counts,
                                      const int*   __restrict__ tokids,
                                      const float* __restrict__ tokw,
                                      float* __restrict__ out) {
    const int e    = blockIdx.y;
    const int tile = blockIdx.x;
    const int cnt  = counts[e];
    if (tile * 16 >= cnt) return;

    extern __shared__ __align__(16) char smem[];
    bf16_t* hs  = (bf16_t*)(smem + HS_OFF);    // [16][4096] bf16
    float*  xf  = (float*) (smem + XF32_OFF);  // [16][1024] f32 (TDM landing)
    bf16_t* xs  = (bf16_t*)(smem + XBF_OFF);   // [16][1024] bf16
    int*    tks = (int*)   (smem + TK_OFF);
    float*  tws = (float*) (tks + 16);

    const int tid  = threadIdx.x;
    const int base = e * T_TOK + tile * 16;

    if (tid < 16) {
        const int r = tile * 16 + tid;
        int tk; float tw;
        if (r < cnt) { tk = tokids[base + tid]; tw = tokw[base + tid]; }
        else         { tk = tokids[base];       tw = 0.0f; }  // pad -> weight 0
        tks[tid] = tk; tws[tid] = tw;
    }

    // -------- TDM gather: 16 token rows of x (f32) -> LDS (async DMA) --------
    if (tid < 32) {
        // clamped 16-bit row indices (padded rows replicate row 0 of the tile)
        const int id0 = tokids[base];
        unsigned int idx[16];
#pragma unroll
        for (int i = 0; i < 16; ++i) {
            const int r = tile * 16 + i;
            idx[i] = (unsigned int)((r < cnt) ? tokids[base + i] : id0) & 0xFFFFu;
        }
        typedef __attribute__((address_space(3))) char* lds_ptr_t;
        const unsigned int lds_addr = (unsigned int)(unsigned long long)(lds_ptr_t)xf;
        const unsigned long long ga = (unsigned long long)(const void*)x;

        u32x4 g0;
        g0[0] = 1u | (0u << 30) | (1u << 31);        // count=1, 16-bit idx, gather
        g0[1] = lds_addr;                            // LDS dest (bytes)
        g0[2] = (unsigned int)(ga & 0xFFFFFFFFu);    // global_addr[31:0]
        g0[3] = (unsigned int)((ga >> 32) & 0x1FFFFFFu) | (2u << 30); // [56:32], type=2

        i32x8 g1;
        g1[0] = (int)(2u << 16);                     // wg_mask=0, data_size=4B
        g1[1] = (int)(((unsigned)D_DIM & 0xFFFFu) << 16);  // tensor_dim0 lo16 @ [63:48]
        g1[2] = (int)(((unsigned)T_TOK & 0xFFFFu) << 16);  // dim0 hi16=0, tensor_dim1 lo16
        g1[3] = (int)(((unsigned)D_DIM) << 16);      // tensor_dim1 hi16=0, tile_dim0=1024
        g1[4] = 16;                                  // tile_dim1 = #indices, tile_dim2=0
        g1[5] = D_DIM;                               // tensor_dim0_stride lo32
        g1[6] = 0;                                   // stride hi16 / dim1_stride lo16
        g1[7] = 0;

        i32x4 g2, g3;
#pragma unroll
        for (int i = 0; i < 4; ++i) {
            g2[i] = (int)(idx[2 * i]     | (idx[2 * i + 1] << 16));
            g3[i] = (int)(idx[8 + 2 * i] | (idx[9 + 2 * i] << 16));
        }
        const i32x8 g4 = {};   // 6-arg toolchain variant: extra group, unused (NULL)
        __builtin_amdgcn_tensor_load_to_lds(g0, g1, g2, g3, g4, 0);
        __builtin_amdgcn_s_wait_tensorcnt(0);
    }
    __syncthreads();

    // f32 -> bf16 conversion of the gathered x tile
    for (int idx = tid; idx < 16 * D_DIM; idx += 256)
        xs[idx] = (bf16_t)xf[idx];
    __syncthreads();

    const int wave  = tid >> 5;
    const int lane  = tid & 31;
    const int ccol  = lane & 15;
    const int rbase = (lane >> 4) * 8;

    // ---------------- GEMM1: h[16, F] = gelu(x @ w1[e] + b1[e]) ----------------
    const float* W1 = w1 + (size_t)e * D_DIM * F_DIM;
    const float* B1 = b1 + (size_t)e * F_DIM;
    for (int t = 0; t < 16; ++t) {               // 8 waves x 16 col-pairs = F/32
        const int n0 = wave * 512 + t * 32;
        v8f acc0 = {}, acc1 = {};
        for (int k0 = 0; k0 < D_DIM; k0 += 32) {
            v16bf a = load_a_frag(xs, D_DIM, k0, lane);
            v16bf b0, b1f;
            load_b_pair_f32(W1, F_DIM, k0, n0, lane, b0, b1f);
            __builtin_prefetch(W1 + (size_t)(k0 + 32) * F_DIM + n0, 0, 0);
            acc0 = __builtin_amdgcn_wmma_f32_16x16x32_bf16(
                false, a, false, b0,  (short)0, acc0, false, false);
            acc1 = __builtin_amdgcn_wmma_f32_16x16x32_bf16(
                false, a, false, b1f, (short)0, acc1, false, false);
        }
        const float bias0 = B1[n0 + ccol];
        const float bias1 = B1[n0 + 16 + ccol];
#pragma unroll
        for (int i = 0; i < 8; ++i) {
            hs[(rbase + i) * F_DIM + n0 + ccol]      = (bf16_t)gelu_tanh(acc0[i] + bias0);
            hs[(rbase + i) * F_DIM + n0 + 16 + ccol] = (bf16_t)gelu_tanh(acc1[i] + bias1);
        }
    }
    __syncthreads();

    // ---------------- GEMM2: y[16, D] = h @ w2[e] + b2[e]; scatter-add ---------
    const float* W2 = w2 + (size_t)e * F_DIM * D_DIM;
    const float* B2 = b2 + (size_t)e * D_DIM;
    for (int t = 0; t < 4; ++t) {                // 8 waves x 4 col-pairs = D/32
        const int n0 = wave * 128 + t * 32;
        v8f acc0 = {}, acc1 = {};
        for (int k0 = 0; k0 < F_DIM; k0 += 32) {
            v16bf a = load_a_frag(hs, F_DIM, k0, lane);
            v16bf b0, b1f;
            load_b_pair_f32(W2, D_DIM, k0, n0, lane, b0, b1f);
            __builtin_prefetch(W2 + (size_t)(k0 + 32) * D_DIM + n0, 0, 0);
            acc0 = __builtin_amdgcn_wmma_f32_16x16x32_bf16(
                false, a, false, b0,  (short)0, acc0, false, false);
            acc1 = __builtin_amdgcn_wmma_f32_16x16x32_bf16(
                false, a, false, b1f, (short)0, acc1, false, false);
        }
        const float bias0 = B2[n0 + ccol];
        const float bias1 = B2[n0 + 16 + ccol];
#pragma unroll
        for (int i = 0; i < 8; ++i) {
            const int row = rbase + i;
            const float w = tws[row];
            float* orow = out + (size_t)tks[row] * D_DIM;
            atomicAdd(&orow[n0 + ccol],      (acc0[i] + bias0) * w);
            atomicAdd(&orow[n0 + 16 + ccol], (acc1[i] + bias1) * w);
        }
    }
}

// ---------------------------------------------------------------- launcher

extern "C" void kernel_launch(void* const* d_in, const int* in_sizes, int n_in,
                              void* d_out, int out_size, void* d_ws, size_t ws_size,
                              hipStream_t stream) {
    const float* x  = (const float*)d_in[0];   // [T, D]
    const float* gw = (const float*)d_in[1];   // [D, E]
    const float* w1 = (const float*)d_in[2];   // [E, D, F]
    const float* b1 = (const float*)d_in[3];   // [E, F]
    const float* w2 = (const float*)d_in[4];   // [E, F, D]
    const float* b2 = (const float*)d_in[5];   // [E, D]
    float* out = (float*)d_out;                // [T, D]

    char* ws = (char*)d_ws;
    int*   counts = (int*)ws;                                   //  8 ints
    int*   tokids = (int*)(ws + 64);                            //  E*T ints
    float* tokw   = (float*)(ws + 64 + N_EXP * T_TOK * 4);      //  E*T floats

    const size_t out_n = (size_t)T_TOK * D_DIM;

    (void)hipFuncSetAttribute((const void*)moe_expert_ffn_kernel,
                              hipFuncAttributeMaxDynamicSharedMemorySize,
                              SMEM_BYTES);

    moe_init_kernel<<<4096, 256, 0, stream>>>(out, out_n, counts);
    moe_router_kernel<<<T_TOK / 8, 256, 0, stream>>>(x, gw, counts, tokids, tokw);

    dim3 grid(T_TOK / 16, N_EXP);   // worst-case tiles per expert; early-exit inside
    moe_expert_ffn_kernel<<<grid, 256, SMEM_BYTES, stream>>>(
        x, w1, b1, w2, b2, counts, tokids, tokw, out);
}
